// miniGPT_62663572848866
// MI455X (gfx1250) — matrix operand order repaired and checked by reading, not compiled, not run on hardware
//
#include <hip/hip_runtime.h>
#include <hip/hip_bf16.h>

// ---------------------------------------------------------------------------
// miniGPT forward on gfx1250 (MI455X).
//  - All GEMMs: v_wmma_f32_16x16x32_bf16 (fp32 accum), operands staged into
//    LDS by the Tensor Data Mover (tensor_load_to_lds, TENSORcnt-tracked,
//    double buffered), fragments read from LDS with ds loads.
//  - Attention: flash-style streaming (no TxT tensor), WMMA for QK^T and PV,
//    LDS round-trip to reshape the score tile C/D layout -> A layout.
// ---------------------------------------------------------------------------

typedef __bf16 v16bf __attribute__((ext_vector_type(16)));
typedef __bf16 v8bf  __attribute__((ext_vector_type(8)));
typedef float  v8f   __attribute__((ext_vector_type(8)));
typedef unsigned int v4u __attribute__((ext_vector_type(4)));
typedef int v8i __attribute__((ext_vector_type(8)));
typedef int v4i __attribute__((ext_vector_type(4)));

#define GPT_B 2
#define GPT_T 2048
#define GPT_C 768
#define GPT_H 12
#define GPT_D 64
#define GPT_L 6
#define GPT_V 50257
#define ROWS (GPT_B * GPT_T)   // 4096

#if defined(__HIP_DEVICE_COMPILE__) && __has_builtin(__builtin_amdgcn_tensor_load_to_lds) && __has_builtin(__builtin_amdgcn_s_wait_tensorcnt)
#define HAVE_TDM 1
#else
#define HAVE_TDM 0
#endif

__device__ inline v8f wmma_bf16(v16bf a, v16bf b, v8f c) {
  // (neg_a, A, neg_b, B, c_mod, C, reuse_a, reuse_b)
  return __builtin_amdgcn_wmma_f32_16x16x32_bf16(false, a, false, b, (short)0, c, false, false);
}

__device__ inline v16bf zero_v16bf() {
  v16bf v;
#pragma unroll
  for (int i = 0; i < 16; ++i) v[i] = (__bf16)0.0f;
  return v;
}
__device__ inline v8f zero_v8f() {
  v8f v;
#pragma unroll
  for (int i = 0; i < 8; ++i) v[i] = 0.0f;
  return v;
}

#if HAVE_TDM
// ---------------------------------------------------------------------------
// Tensor Data Mover: 2-D bf16 tile load Global -> LDS.
// D# packing per CDNA5 ISA 8.3/8.4:
//  group0: count=1 [1:0]; lds_addr [63:32]; global_addr [120:64]; type=2 [127:126]
//  group1: data_size [17:16] (=1: 2 bytes); tensor_dim0 [79:48];
//          tensor_dim1 [111:80]; tile_dim0 [127:112]; tile_dim1 [143:128];
//          tensor_dim0_stride [207:160] (element units)
// Rows y >= tensor_dim1 are zero-filled by the TDM (OOB read -> 0), which
// handles the ragged LM-head edge (N = 50257).
// This toolchain exposes the 6-arg builtin:
//  (uint32x4 g0, int32x8 g1, int32x4 g2, int32x4 g3, int32x8 extra, i32 cpol)
// Groups 2/3 unused for 2-D tiles -> zero-filled.
// ---------------------------------------------------------------------------
__device__ inline void tdm_load_2d_bf16(unsigned lds_off, const void* gptr,
                                        unsigned tensor_d0, unsigned tensor_d1,
                                        unsigned tile_d0, unsigned tile_d1,
                                        unsigned long long stride0_elems) {
  const unsigned long long ga = (unsigned long long)gptr;
  v4u g0;
  g0[0] = 1u;                                           // count=1 (user mode)
  g0[1] = lds_off;                                      // lds_addr (bytes)
  g0[2] = (unsigned)(ga & 0xFFFFFFFFu);                 // global_addr lo
  g0[3] = (unsigned)((ga >> 32) & 0x01FFFFFFu) | (2u << 30);  // hi + type=2
  v8i g1;
  g1[0] = (int)(1u << 16);                              // data_size = 2 bytes
  g1[1] = (int)((tensor_d0 & 0xFFFFu) << 16);           // tensor_dim0 lo16
  g1[2] = (int)((tensor_d0 >> 16) | ((tensor_d1 & 0xFFFFu) << 16));
  g1[3] = (int)((tensor_d1 >> 16) | (tile_d0 << 16));   // tile_dim0
  g1[4] = (int)(tile_d1 & 0xFFFFu);                     // tile_dim1 (tile_dim2=0)
  g1[5] = (int)(stride0_elems & 0xFFFFFFFFull);         // dim0 stride lo32
  g1[6] = (int)((stride0_elems >> 32) & 0xFFFFull);     // dim0 stride hi16
  g1[7] = 0;                                            // dim1 stride (2-D)
  const v4i z4 = {0, 0, 0, 0};
  const v8i z8 = {0, 0, 0, 0, 0, 0, 0, 0};
  __builtin_amdgcn_tensor_load_to_lds(g0, g1, z4, z4, z8, 0);
}
__device__ inline unsigned lds_off_u32(const void* p) {
  return (unsigned)(unsigned long long)p;   // flat LDS addr: offset in [31:0]
}
#endif

// ---------------------------------------------------------------------------
// fp32 -> bf16 conversion (weights, activations)
// ---------------------------------------------------------------------------
__global__ void cvt_f32_bf16(const float* __restrict__ in, __bf16* __restrict__ out, size_t n) {
  size_t i = (size_t)blockIdx.x * blockDim.x + threadIdx.x;
  size_t stride = (size_t)gridDim.x * blockDim.x;
  for (; i < n; i += stride) out[i] = (__bf16)in[i];
}

// ---------------------------------------------------------------------------
// Embedding: h[row, :] = wte[x[row], :] + wpe[row % T, :]
// ---------------------------------------------------------------------------
__global__ void embed_kernel(const int* __restrict__ x, const float* __restrict__ wte,
                             const float* __restrict__ wpe, float* __restrict__ h) {
  const int row = blockIdx.x;            // 0..4095
  const int t = row & (GPT_T - 1);
  const int id = x[row];
  const float* we = wte + (size_t)id * GPT_C;
  const float* wp = wpe + (size_t)t * GPT_C;
  float* hr = h + (size_t)row * GPT_C;
  for (int c = threadIdx.x; c < GPT_C; c += blockDim.x) hr[c] = we[c] + wp[c];
}

// ---------------------------------------------------------------------------
// LayerNorm (fp32 in) -> bf16 out. One block (256 thr = 8 waves) per row.
// ---------------------------------------------------------------------------
__global__ __launch_bounds__(256)
void layernorm_bf16(const float* __restrict__ x, const float* __restrict__ w,
                    const float* __restrict__ b, __bf16* __restrict__ out) {
  __shared__ float redA[8];
  __shared__ float redB[8];
  const int row = blockIdx.x;
  const int tid = threadIdx.x;
  const float* xr = x + (size_t)row * GPT_C;
  const float v0 = xr[tid], v1 = xr[tid + 256], v2 = xr[tid + 512];
  float s = v0 + v1 + v2;
  float ss = v0 * v0 + v1 * v1 + v2 * v2;
#pragma unroll
  for (int off = 16; off; off >>= 1) {
    s += __shfl_xor(s, off, 32);
    ss += __shfl_xor(ss, off, 32);
  }
  if ((tid & 31) == 0) { redA[tid >> 5] = s; redB[tid >> 5] = ss; }
  __syncthreads();
  float ts = 0.f, tss = 0.f;
#pragma unroll
  for (int i = 0; i < 8; ++i) { ts += redA[i]; tss += redB[i]; }
  const float mean = ts * (1.0f / GPT_C);
  const float var = tss * (1.0f / GPT_C) - mean * mean;
  const float rs = rsqrtf(var + 1e-5f);
  __bf16* orow = out + (size_t)row * GPT_C;
  orow[tid]       = (__bf16)((v0 - mean) * rs * w[tid]       + b[tid]);
  orow[tid + 256] = (__bf16)((v1 - mean) * rs * w[tid + 256] + b[tid + 256]);
  orow[tid + 512] = (__bf16)((v2 - mean) * rs * w[tid + 512] + b[tid + 512]);
}

// ---------------------------------------------------------------------------
// Fallback fragment loads straight from global (non-TDM toolchains/host pass).
// A layout: lane(0..15)->M; elem i -> K = (lane>>4)*16 + i.
// B from row-major W[N,K]: lane->N; elem i -> K = (lane>>4)*16 + i.
// ---------------------------------------------------------------------------
__device__ inline v16bf load_a_frag_g(const __bf16* __restrict__ A, int m0, int k0,
                                      int lda, int lane) {
  const int m = m0 + (lane & 15);
  const int kb = lane >> 4;
  const __bf16* p = A + (size_t)m * lda + k0 + kb * 16;
  const v8bf lo = *(const v8bf*)(p);
  const v8bf hi = *(const v8bf*)(p + 8);
  v16bf a;
#pragma unroll
  for (int i = 0; i < 8; ++i) { a[i] = lo[i]; a[i + 8] = hi[i]; }
  return a;
}
__device__ inline v16bf load_b_frag_g(const __bf16* __restrict__ W, int n0, int k0,
                                      int ldb, int N, int lane) {
  const int n = n0 + (lane & 15);
  const int kb = lane >> 4;
  if (n >= N) return zero_v16bf();
  return *(const v16bf*)(W + (size_t)n * ldb + k0 + kb * 16);
}

// ---------------------------------------------------------------------------
// Generic GEMM: out[M,N] (+)= act(A[M,K] @ W[N,K]^T + bias)
// Block: 256 thr = 8 waves in 2(M) x 4(N); block tile 64x128; wave tile 32x32.
// TDM path: wave 0 DMAs the 64x32 A tile and 128x32 B tile into double-
// buffered LDS (tensor_load_to_lds), waits on TENSORcnt, block barriers, and
// all waves read WMMA fragments from LDS (ds loads). No per-lane global
// address math in the hot loop.
// ---------------------------------------------------------------------------
template <bool HAS_BIAS, bool RELU, bool OUT_BF16, bool ACCUM>
__global__ __launch_bounds__(256, 2)
void gemm_xwT(const __bf16* __restrict__ A, const __bf16* __restrict__ W,
              const float* __restrict__ bias, void* __restrict__ outp,
              int M, int N, int K, int lda, int ldb, int ldo) {
  const int lane = threadIdx.x & 31;
  const int w = threadIdx.x >> 5;
  const int wm = w >> 2;          // 0..1
  const int wn = w & 3;           // 0..3
  const int m_base = blockIdx.y * 64;
  const int n_base = blockIdx.x * 128;
  const int lcol = lane & 15;
  const int lhalf = lane >> 4;

  v8f acc00 = zero_v8f(), acc01 = zero_v8f(), acc10 = zero_v8f(), acc11 = zero_v8f();
  const int nk = K / 32;

#if HAVE_TDM
  __shared__ __attribute__((aligned(128))) __bf16 sA[2][64 * 32];
  __shared__ __attribute__((aligned(128))) __bf16 sB[2][128 * 32];
  const unsigned nrem = (unsigned)((N - n_base) < 128 ? (N - n_base) : 128);

  if (w == 0) {  // wave 0 drives the TDM (EXEC-independent, per-wave op)
    tdm_load_2d_bf16(lds_off_u32(&sA[0][0]), A + (size_t)m_base * lda,
                     32u, 64u, 32u, 64u, (unsigned long long)lda);
    tdm_load_2d_bf16(lds_off_u32(&sB[0][0]), W + (size_t)n_base * ldb,
                     32u, nrem, 32u, 128u, (unsigned long long)ldb);
  }
  for (int k = 0; k < nk; ++k) {
    const int cur = k & 1;
    if (w == 0) {
      if (k + 1 < nk) {
        const int k0n = (k + 1) * 32;
        tdm_load_2d_bf16(lds_off_u32(&sA[cur ^ 1][0]), A + (size_t)m_base * lda + k0n,
                         32u, 64u, 32u, 64u, (unsigned long long)lda);
        tdm_load_2d_bf16(lds_off_u32(&sB[cur ^ 1][0]), W + (size_t)n_base * ldb + k0n,
                         32u, nrem, 32u, 128u, (unsigned long long)ldb);
        __builtin_amdgcn_s_wait_tensorcnt((short)2);  // step k landed; k+1 in flight
      } else {
        __builtin_amdgcn_s_wait_tensorcnt((short)0);
      }
    }
    __syncthreads();  // LDS tiles for step k visible to all waves

    v16bf a0, a1, b0, b1;
    {
      const __bf16* ap = &sA[cur][(wm * 32 + lcol) * 32 + lhalf * 16];
      v8bf lo = *(const v8bf*)(ap);
      v8bf hi = *(const v8bf*)(ap + 8);
#pragma unroll
      for (int i = 0; i < 8; ++i) { a0[i] = lo[i]; a0[i + 8] = hi[i]; }
      ap += 16 * 32;
      lo = *(const v8bf*)(ap);
      hi = *(const v8bf*)(ap + 8);
#pragma unroll
      for (int i = 0; i < 8; ++i) { a1[i] = lo[i]; a1[i + 8] = hi[i]; }
    }
    b0 = *(const v16bf*)(&sB[cur][(wn * 32 + lcol) * 32 + lhalf * 16]);
    b1 = *(const v16bf*)(&sB[cur][(wn * 32 + 16 + lcol) * 32 + lhalf * 16]);

    acc00 = wmma_bf16(a0, b0, acc00);
    acc01 = wmma_bf16(a0, b1, acc01);
    acc10 = wmma_bf16(a1, b0, acc10);
    acc11 = wmma_bf16(a1, b1, acc11);

    __syncthreads();  // reads of buffer `cur` done before TDM reuses it
  }
#else
  const int mb = m_base + wm * 32;
  const int nb = n_base + wn * 32;
  for (int k = 0; k < nk; ++k) {
    const int k0 = k * 32;
    if (k0 + 32 < K) {
      __builtin_prefetch(A + (size_t)(mb + lcol) * lda + k0 + 32, 0, 1);
      __builtin_prefetch(W + (size_t)(nb + lcol) * ldb + k0 + 32, 0, 1);
    }
    const v16bf a0 = load_a_frag_g(A, mb, k0, lda, lane);
    const v16bf a1 = load_a_frag_g(A, mb + 16, k0, lda, lane);
    const v16bf b0 = load_b_frag_g(W, nb, k0, ldb, N, lane);
    const v16bf b1 = load_b_frag_g(W, nb + 16, k0, ldb, N, lane);
    acc00 = wmma_bf16(a0, b0, acc00);
    acc01 = wmma_bf16(a0, b1, acc01);
    acc10 = wmma_bf16(a1, b0, acc10);
    acc11 = wmma_bf16(a1, b1, acc11);
  }
#endif

  // C/D layout: VGPR r, lane(0..15)->N=col, lane(16..31)->M += 8.
  const v8f accs[4] = {acc00, acc01, acc10, acc11};
#pragma unroll
  for (int mt = 0; mt < 2; ++mt) {
#pragma unroll
    for (int nt = 0; nt < 2; ++nt) {
      const v8f acc = accs[mt * 2 + nt];
      const int col = n_base + wn * 32 + nt * 16 + lcol;
      if (col >= N) continue;
      const float bv = HAS_BIAS ? bias[col] : 0.0f;
      const int row0 = m_base + wm * 32 + mt * 16 + 8 * lhalf;
#pragma unroll
      for (int r = 0; r < 8; ++r) {
        float v = acc[r] + bv;
        if (RELU) v = fmaxf(v, 0.0f);
        const size_t idx = (size_t)(row0 + r) * ldo + col;
        if (OUT_BF16)      ((__bf16*)outp)[idx] = (__bf16)v;
        else if (ACCUM)    ((float*)outp)[idx] += v;
        else               ((float*)outp)[idx] = v;
      }
    }
  }
}

// ---------------------------------------------------------------------------
// Flash attention. qkv fp32 [4096, 2304]; adds attention output into h.
// Grid: (T/64, B*H); block 128 = 4 waves; each wave owns 16 query rows.
// ---------------------------------------------------------------------------
__global__ __launch_bounds__(128, 2)
void attention_kernel(const float* __restrict__ qkv, float* __restrict__ h) {
  __shared__ __bf16 sP[4][16 * 32];
  const int lane = threadIdx.x & 31;
  const int w = threadIdx.x >> 5;
  const int bh = blockIdx.y;
  const int b = bh / GPT_H;
  const int hh = bh % GPT_H;
  const int q0 = blockIdx.x * 64 + w * 16;
  const size_t rowb = (size_t)b * GPT_T;
  const int lcol = lane & 15;
  const int lhalf = lane >> 4;

  const float* qbase = qkv + rowb * (3 * GPT_C) + hh * GPT_D;
  const float* kbase = qbase + GPT_C;
  const float* vbase = qbase + 2 * GPT_C;

  // Q fragments (two K=32 steps over D=64), pre-scaled by 1/sqrt(D)=0.125
  v16bf qa[2];
#pragma unroll
  for (int ks = 0; ks < 2; ++ks) {
    const float* qr = qbase + (size_t)(q0 + lcol) * (3 * GPT_C) + ks * 32 + lhalf * 16;
#pragma unroll
    for (int i = 0; i < 16; ++i) {
      const int d = i;
      qa[ks][i] = (__bf16)(qr[d] * 0.125f);
    }
  }

  v8f o[4] = {zero_v8f(), zero_v8f(), zero_v8f(), zero_v8f()};
  float mrow[8], lrow[8];
#pragma unroll
  for (int r = 0; r < 8; ++r) { mrow[r] = -1e30f; lrow[r] = 0.0f; }

  for (int s0 = 0; s0 < q0 + 16; s0 += 32) {
    // ---- S = QK^T for 32 keys (two 16-key WMMA column tiles) ----
    v8f sa = zero_v8f(), sb = zero_v8f();
#pragma unroll
    for (int ks = 0; ks < 2; ++ks) {
      v16bf kf0, kf1;
      const float* kr0 = kbase + (size_t)(s0 + lcol) * (3 * GPT_C) + ks * 32 + lhalf * 16;
      const float* kr1 = kr0 + (size_t)16 * (3 * GPT_C);
#pragma unroll
      for (int i = 0; i < 16; ++i) {
        kf0[i] = (__bf16)kr0[i];
        kf1[i] = (__bf16)kr1[i];
      }
      sa = wmma_bf16(qa[ks], kf0, sa);
      sb = wmma_bf16(qa[ks], kf1, sb);
    }

    // ---- causal mask + online softmax (row = r + 8*lhalf spans 16 lanes) ----
    float scale_old[8], p0[8], p1[8];
#pragma unroll
    for (int r = 0; r < 8; ++r) {
      const int tq = q0 + r + 8 * lhalf;
      const int key0 = s0 + lcol;
      const int key1 = key0 + 16;
      const float sv0 = (key0 <= tq) ? sa[r] : -1e30f;
      const float sv1 = (key1 <= tq) ? sb[r] : -1e30f;
      float vmax = fmaxf(sv0, sv1);
      vmax = fmaxf(vmax, __shfl_xor(vmax, 1, 32));
      vmax = fmaxf(vmax, __shfl_xor(vmax, 2, 32));
      vmax = fmaxf(vmax, __shfl_xor(vmax, 4, 32));
      vmax = fmaxf(vmax, __shfl_xor(vmax, 8, 32));
      const float mnew = fmaxf(mrow[r], vmax);
      scale_old[r] = __expf(mrow[r] - mnew);
      mrow[r] = mnew;
      p0[r] = __expf(sv0 - mnew);
      p1[r] = __expf(sv1 - mnew);
      float ls = p0[r] + p1[r];
      ls += __shfl_xor(ls, 1, 32);
      ls += __shfl_xor(ls, 2, 32);
      ls += __shfl_xor(ls, 4, 32);
      ls += __shfl_xor(ls, 8, 32);
      lrow[r] = lrow[r] * scale_old[r] + ls;
    }

    // ---- rescale running O ----
#pragma unroll
    for (int nt = 0; nt < 4; ++nt)
#pragma unroll
      for (int r = 0; r < 8; ++r) o[nt][r] *= scale_old[r];

    // ---- stage P (bf16, row-major 16x32) through LDS: C/D -> A layout ----
#pragma unroll
    for (int r = 0; r < 8; ++r) {
      sP[w][(r + 8 * lhalf) * 32 + lcol]      = (__bf16)p0[r];
      sP[w][(r + 8 * lhalf) * 32 + 16 + lcol] = (__bf16)p1[r];
    }
    asm volatile("s_wait_dscnt 0" ::: "memory");
    v16bf pa;
#pragma unroll
    for (int i = 0; i < 16; ++i) {
      const int kk = lhalf * 16 + i;
      pa[i] = sP[w][lcol * 32 + kk];
    }
    asm volatile("" ::: "memory");

    // ---- O += P @ V (V as B operand: K=key, N=feature; 4 column tiles) ----
#pragma unroll
    for (int nt = 0; nt < 4; ++nt) {
      v16bf vf;
      const float* vr = vbase + (size_t)(s0 + lhalf * 16) * (3 * GPT_C) + nt * 16 + lcol;
#pragma unroll
      for (int i = 0; i < 16; ++i) vf[i] = (__bf16)vr[(size_t)i * (3 * GPT_C)];
      o[nt] = wmma_bf16(pa, vf, o[nt]);
    }
  }

  // ---- epilogue: h += O / l  (residual add, each element owned uniquely) ----
#pragma unroll
  for (int nt = 0; nt < 4; ++nt) {
#pragma unroll
    for (int r = 0; r < 8; ++r) {
      const size_t row = rowb + q0 + r + 8 * lhalf;
      const int col = hh * GPT_D + nt * 16 + lcol;
      h[row * GPT_C + col] += o[nt][r] / lrow[r];
    }
  }
}

// ---------------------------------------------------------------------------
// Loss: per-row log_softmax + NLL, mean via atomics.
// ---------------------------------------------------------------------------
__global__ void zero_loss(float* p) { if (threadIdx.x == 0 && blockIdx.x == 0) *p = 0.0f; }

__global__ __launch_bounds__(256)
void loss_kernel(const float* __restrict__ logits, const int* __restrict__ target,
                 float* __restrict__ loss) {
  __shared__ float red[8];
  const int row = blockIdx.x;
  const int tid = threadIdx.x;
  const float* lr = logits + (size_t)row * GPT_V;

  float mx = -1e30f;
  for (int c = tid; c < GPT_V; c += 256) mx = fmaxf(mx, lr[c]);
#pragma unroll
  for (int off = 16; off; off >>= 1) mx = fmaxf(mx, __shfl_xor(mx, off, 32));
  if ((tid & 31) == 0) red[tid >> 5] = mx;
  __syncthreads();
  mx = red[0];
#pragma unroll
  for (int i = 1; i < 8; ++i) mx = fmaxf(mx, red[i]);
  __syncthreads();

  float s = 0.0f;
  for (int c = tid; c < GPT_V; c += 256) s += __expf(lr[c] - mx);
#pragma unroll
  for (int off = 16; off; off >>= 1) s += __shfl_xor(s, off, 32);
  if ((tid & 31) == 0) red[tid >> 5] = s;
  __syncthreads();
  if (tid == 0) {
    float tot = 0.0f;
#pragma unroll
    for (int i = 0; i < 8; ++i) tot += red[i];
    const float lp = lr[target[row]] - mx - logf(tot);
    atomicAdd(loss, -lp * (1.0f / (float)ROWS));
  }
}

// ---------------------------------------------------------------------------
// Host orchestration
// ---------------------------------------------------------------------------
extern "C" void kernel_launch(void* const* d_in, const int* in_sizes, int n_in,
                              void* d_out, int out_size, void* d_ws, size_t ws_size,
                              hipStream_t stream) {
  (void)in_sizes; (void)n_in; (void)out_size; (void)ws_size;

  const int*   x      = (const int*)  d_in[0];
  const int*   target = (const int*)  d_in[1];
  const float* wte    = (const float*)d_in[2];
  const float* wpe    = (const float*)d_in[3];
  const float* ln1_w  = (const float*)d_in[4];
  const float* ln1_b  = (const float*)d_in[5];
  const float* attn_w = (const float*)d_in[6];
  const float* attn_b = (const float*)d_in[7];
  const float* ln2_w  = (const float*)d_in[8];
  const float* ln2_b  = (const float*)d_in[9];
  const float* ff1_w  = (const float*)d_in[10];
  const float* ff1_b  = (const float*)d_in[11];
  const float* ff2_w  = (const float*)d_in[12];
  const float* ff2_b  = (const float*)d_in[13];
  const float* lm_w   = (const float*)d_in[14];

  float* loss   = (float*)d_out;
  float* logits = (float*)d_out + 1;

  // Workspace carve-up (256B aligned blocks)
  char* ws = (char*)d_ws;
  size_t off = 0;
  auto carve = [&](size_t bytes) -> void* {
    void* p = ws + off;
    off += (bytes + 255) & ~(size_t)255;
    return p;
  };
  float*  h       = (float*) carve((size_t)ROWS * GPT_C * 4);
  __bf16* xnb     = (__bf16*)carve((size_t)ROWS * GPT_C * 2);
  float*  qkvb    = (float*) carve((size_t)ROWS * 3 * GPT_C * 4);
  __bf16* ffab    = (__bf16*)carve((size_t)ROWS * 4 * GPT_C * 2);
  __bf16* attn_wb = (__bf16*)carve((size_t)GPT_L * 3 * GPT_C * GPT_C * 2);
  __bf16* ff1_wb  = (__bf16*)carve((size_t)GPT_L * 4 * GPT_C * GPT_C * 2);
  __bf16* ff2_wb  = (__bf16*)carve((size_t)GPT_L * GPT_C * 4 * GPT_C * 2);
  __bf16* lm_wb   = (__bf16*)carve((size_t)GPT_V * GPT_C * 2);

  // One-time weight conversion fp32 -> bf16
  cvt_f32_bf16<<<4096, 256, 0, stream>>>(attn_w, attn_wb, (size_t)GPT_L * 3 * GPT_C * GPT_C);
  cvt_f32_bf16<<<4096, 256, 0, stream>>>(ff1_w,  ff1_wb,  (size_t)GPT_L * 4 * GPT_C * GPT_C);
  cvt_f32_bf16<<<4096, 256, 0, stream>>>(ff2_w,  ff2_wb,  (size_t)GPT_L * GPT_C * 4 * GPT_C);
  cvt_f32_bf16<<<4096, 256, 0, stream>>>(lm_w,   lm_wb,   (size_t)GPT_V * GPT_C);

  embed_kernel<<<ROWS, 256, 0, stream>>>(x, wte, wpe, h);

  const dim3 gQKV(3 * GPT_C / 128, ROWS / 64);   // (18, 64)
  const dim3 gFF1(4 * GPT_C / 128, ROWS / 64);   // (24, 64)
  const dim3 gFF2(GPT_C / 128, ROWS / 64);       // (6, 64)
  const dim3 gLM((GPT_V + 127) / 128, ROWS / 64);// (393, 64)
  const dim3 gATT(GPT_T / 64, GPT_B * GPT_H);    // (32, 24)

  for (int l = 0; l < GPT_L; ++l) {
    layernorm_bf16<<<ROWS, 256, 0, stream>>>(h, ln1_w + l * GPT_C, ln1_b + l * GPT_C, xnb);
    // qkv = xn @ attn_w[l]^T + attn_b[l]
    gemm_xwT<true, false, false, false><<<gQKV, 256, 0, stream>>>(
        xnb, attn_wb + (size_t)l * 3 * GPT_C * GPT_C, attn_b + (size_t)l * 3 * GPT_C,
        qkvb, ROWS, 3 * GPT_C, GPT_C, GPT_C, GPT_C, 3 * GPT_C);
    // h += attention(qkv)
    attention_kernel<<<gATT, 128, 0, stream>>>(qkvb, h);
    layernorm_bf16<<<ROWS, 256, 0, stream>>>(h, ln2_w + l * GPT_C, ln2_b + l * GPT_C, xnb);
    // ffa = relu(xn @ ff1_w[l]^T + ff1_b[l])   (bf16 out)
    gemm_xwT<true, true, true, false><<<gFF1, 256, 0, stream>>>(
        xnb, ff1_wb + (size_t)l * 4 * GPT_C * GPT_C, ff1_b + (size_t)l * 4 * GPT_C,
        ffab, ROWS, 4 * GPT_C, GPT_C, GPT_C, GPT_C, 4 * GPT_C);
    // h += ffa @ ff2_w[l]^T + ff2_b[l]
    gemm_xwT<true, false, false, true><<<gFF2, 256, 0, stream>>>(
        ffab, ff2_wb + (size_t)l * GPT_C * 4 * GPT_C, ff2_b + (size_t)l * GPT_C,
        h, ROWS, GPT_C, 4 * GPT_C, 4 * GPT_C, 4 * GPT_C, GPT_C);
  }

  // logits = h @ lm_w^T  (no bias; note reference's ln_f is dead code)
  cvt_f32_bf16<<<4096, 256, 0, stream>>>(h, xnb, (size_t)ROWS * GPT_C);
  gemm_xwT<false, false, false, false><<<gLM, 256, 0, stream>>>(
      xnb, lm_wb, nullptr, logits, ROWS, GPT_V, GPT_C, GPT_C, GPT_C, GPT_V);

  zero_loss<<<1, 1, 0, stream>>>(loss);
  loss_kernel<<<ROWS, 256, 0, stream>>>(logits, target, loss);
}
